// CausalCrisisModel_79250736546290
// MI455X (gfx1250) — matrix operand order repaired and checked by reading, not compiled, not run on hardware
//
#include <hip/hip_runtime.h>
#include <hip/hip_bf16.h>
#include <math.h>

// ---------------------------------------------------------------------------
// Types for CDNA5 WMMA
// ---------------------------------------------------------------------------
typedef __attribute__((ext_vector_type(16))) __bf16 v16bf;
typedef __attribute__((ext_vector_type(8)))  float  v8f;

union FragBF16 {
    __bf16 h[16];
    v16bf  v;
    uint4  q[2];
};

// ---------------------------------------------------------------------------
// f32 -> bf16 weight conversion
// ---------------------------------------------------------------------------
__global__ void cvt_bf16_kernel(const float* __restrict__ in, __bf16* __restrict__ out, long n) {
    long i = (long)blockIdx.x * 256 + threadIdx.x;
    if (i < n) out[i] = (__bf16)in[i];
}

// ---------------------------------------------------------------------------
// Generic GEMM: Y[M,N] = act( A[M,K] @ W[N,K]^T + bias (+ residual) )
// A f32 row-major (leading dim lda), W bf16 [N,K] row-major.
// Block = 256 threads = 8 wave32 waves, each wave owns a 64x64 output tile
// (4x4 WMMA 16x16 tiles -> 16 v_wmma per 32-deep K step, 1 barrier per step,
// double-buffered LDS). Staging (global->bf16->LDS) is written AFTER the WMMA
// section with a short live range: gfx1250 caps direct VGPR addressing at 256,
// and holding staging registers across the WMMAs caused scratch spills. The
// scheduler hoists the dependency-free global loads above the WMMAs only as
// far as register pressure allows; remaining latency is hidden cross-wave.
// Template: BMv x BNv block tile, (BMv/64)*(BNv/64) must be 8.
// Requires M%BMv==0, N%BNv==0, K%32==0 (true for every call below).
// ACT: 0=none, 1=relu, 2=gelu(exact). RES: fused residual add (pitch N).
// ---------------------------------------------------------------------------
#define BK 32
#define LPITCH 40   // bf16 elems per LDS row (80B: 16B-aligned, conflict-free pad)

template<int BMv, int BNv, int ACT, bool RES>
__global__ __launch_bounds__(256)
void gemm_bf16_kernel(const float* __restrict__ A, long lda,
                      const __bf16* __restrict__ W,
                      const float* __restrict__ bias,
                      const float* __restrict__ residual,
                      float* __restrict__ Y,
                      int M, int N, int K)
{
    constexpr int A_ITERS = (BMv * 2) / 256;   // 16-elem chunks per thread for A
    constexpr int B_ITERS = (BNv * 2) / 256;   // 16-elem chunks per thread for B
    constexpr int NWN     = BNv / 64;          // waves along N

    __shared__ __bf16 sm[2][(BMv + BNv) * LPITCH];

    const int tid  = threadIdx.x;
    const int m0   = blockIdx.y * BMv;
    const int n0   = blockIdx.x * BNv;

    const int wid  = tid >> 5;
    const int lane = tid & 31;
    const int wm   = wid / NWN;
    const int wn   = wid % NWN;
    const int lm   = lane & 15;
    const int lh   = lane >> 4;

    v8f zero = {0.f,0.f,0.f,0.f,0.f,0.f,0.f,0.f};
    v8f acc[4][4];
    #pragma unroll
    for (int i = 0; i < 4; i++)
        #pragma unroll
        for (int j = 0; j < 4; j++)
            acc[i][j] = zero;

    // ---- staging assignment: chunk id -> (row, 16-elem segment) ----
    const float*  aPtr[A_ITERS];  unsigned aOff[A_ITERS];
    const __bf16* bPtr[B_ITERS];  unsigned bOff[B_ITERS];
    #pragma unroll
    for (int a = 0; a < A_ITERS; a++) {
        const int id  = tid + 256 * a;
        const int row = id >> 1, seg = (id & 1) << 4;
        aPtr[a] = A + (size_t)(m0 + row) * lda + seg;
        aOff[a] = row * LPITCH + seg;
    }
    #pragma unroll
    for (int b = 0; b < B_ITERS; b++) {
        const int id  = tid + 256 * b;
        const int row = id >> 1, seg = (id & 1) << 4;
        bPtr[b] = W + (size_t)(n0 + row) * K + seg;
        bOff[b] = BMv * LPITCH + row * LPITCH + seg;
    }

    // load global -> convert -> LDS store, all short-lived (no held staging regs)
    auto stage = [&](int kk, int buf) {
        __bf16* base = sm[buf];
        #pragma unroll
        for (int a = 0; a < A_ITERS; a++) {
            const float4* p = (const float4*)(aPtr[a] + kk);
            float4 f0 = p[0], f1 = p[1], f2 = p[2], f3 = p[3];
            FragBF16 t;
            t.h[0]=(__bf16)f0.x;  t.h[1]=(__bf16)f0.y;  t.h[2]=(__bf16)f0.z;  t.h[3]=(__bf16)f0.w;
            t.h[4]=(__bf16)f1.x;  t.h[5]=(__bf16)f1.y;  t.h[6]=(__bf16)f1.z;  t.h[7]=(__bf16)f1.w;
            t.h[8]=(__bf16)f2.x;  t.h[9]=(__bf16)f2.y;  t.h[10]=(__bf16)f2.z; t.h[11]=(__bf16)f2.w;
            t.h[12]=(__bf16)f3.x; t.h[13]=(__bf16)f3.y; t.h[14]=(__bf16)f3.z; t.h[15]=(__bf16)f3.w;
            uint4* dst = (uint4*)&base[aOff[a]];
            dst[0] = t.q[0]; dst[1] = t.q[1];
        }
        #pragma unroll
        for (int b = 0; b < B_ITERS; b++) {
            const uint4* p = (const uint4*)(bPtr[b] + kk);
            uint4 q0 = p[0], q1 = p[1];
            uint4* dst = (uint4*)&base[bOff[b]];
            dst[0] = q0; dst[1] = q1;
        }
    };

    const int nk = K / BK;
    stage(0, 0);
    __syncthreads();

    for (int t = 0; t < nk; ++t) {
        const int  cur      = t & 1;
        const bool haveNext = (t + 1 < nk);

        if (t + 2 < nk) {
            // speculative L2 prefetch of stage t+2 (global_prefetch_b8)
            __builtin_prefetch(aPtr[0] + (t + 2) * BK, 0, 1);
            __builtin_prefetch(bPtr[0] + (t + 2) * BK, 0, 1);
        }

        const __bf16* As = sm[cur];
        const __bf16* Bs = sm[cur] + BMv * LPITCH;

        // ---- fragment loads (per ISA 16-bit A 16x32 / B 32x16 layouts) ----
        FragBF16 afr[4], bfr[4];
        #pragma unroll
        for (int mi = 0; mi < 4; mi++) {
            const __bf16* ap = &As[(wm * 64 + mi * 16 + lm) * LPITCH];
            afr[mi].q[0] = *(const uint4*)(ap + lh * 8);        // K = 8*half + 0..7
            afr[mi].q[1] = *(const uint4*)(ap + 16 + lh * 8);   // K = 16 + 8*half + 0..7
        }
        #pragma unroll
        for (int ni = 0; ni < 4; ni++) {
            const __bf16* bp = &Bs[(wn * 64 + ni * 16 + lm) * LPITCH];
            bfr[ni].q[0] = *(const uint4*)(bp + lh * 8);        // K = 8*half + 0..7
            bfr[ni].q[1] = *(const uint4*)(bp + 16 + lh * 8);   // K = 16 + 8*half + 0..7
        }

        #pragma unroll
        for (int mi = 0; mi < 4; mi++)
            #pragma unroll
            for (int ni = 0; ni < 4; ni++)
                acc[mi][ni] = __builtin_amdgcn_wmma_f32_16x16x32_bf16(
                    false, afr[mi].v, false, bfr[ni].v,
                    (short)0, acc[mi][ni], false, false);

        if (haveNext) {
            stage((t + 1) * BK, cur ^ 1);   // writes buf nxt; reads of cur are done above
            __syncthreads();
        }
    }

    // ---- epilogue (straight-line: templated ACT/RES, incremental addressing) ----
    #pragma unroll
    for (int mi = 0; mi < 4; mi++) {
        #pragma unroll
        for (int ni = 0; ni < 4; ni++) {
            const int col  = n0 + wn * 64 + ni * 16 + lm;
            const int row0 = m0 + wm * 64 + mi * 16 + lh * 8;
            const float bval = bias[col];
            size_t idx = (size_t)row0 * N + col;
            #pragma unroll
            for (int rr = 0; rr < 8; rr++) {
                float v = acc[mi][ni][rr] + bval;
                if constexpr (RES)      v += residual[idx];
                if constexpr (ACT == 1) v = fmaxf(v, 0.f);
                if constexpr (ACT == 2) v = 0.5f * v * (1.f + erff(v * 0.70710678118654752f));
                Y[idx] = v;
                idx += N;
            }
        }
    }
}

// ---------------------------------------------------------------------------
// Row-wise LayerNorm (optionally fused exact GELU). One block (256 thr) / row.
// Row cached in dynamic LDS so in-place use is safe. Strided in/out for
// writing directly into concat buffers.
// ---------------------------------------------------------------------------
__global__ void layernorm_kernel(const float* __restrict__ in, long in_pitch,
                                 const float* __restrict__ g, const float* __restrict__ b,
                                 float* __restrict__ out, long out_pitch,
                                 int d, int do_gelu)
{
    extern __shared__ float smem[];     // d row values + 256 reduction slots
    float* row = smem;
    float* red = smem + d;
    const int tid = threadIdx.x;
    const long r = blockIdx.x;

    const float* ip = in + r * in_pitch;
    float acc = 0.f;
    for (int i = tid; i < d; i += 256) { float v = ip[i]; row[i] = v; acc += v; }
    red[tid] = acc; __syncthreads();
    for (int s = 128; s > 0; s >>= 1) { if (tid < s) red[tid] += red[tid + s]; __syncthreads(); }
    const float mean = red[0] / d;
    __syncthreads();

    float acc2 = 0.f;
    for (int i = tid; i < d; i += 256) { float v = row[i] - mean; acc2 += v * v; }
    red[tid] = acc2; __syncthreads();
    for (int s = 128; s > 0; s >>= 1) { if (tid < s) red[tid] += red[tid + s]; __syncthreads(); }
    const float rstd = rsqrtf(red[0] / d + 1e-5f);

    float* op = out + r * out_pitch;
    for (int i = tid; i < d; i += 256) {
        float v = (row[i] - mean) * rstd * g[i] + b[i];
        if (do_gelu) v = 0.5f * v * (1.f + erff(v * 0.70710678118654752f));
        op[i] = v;
    }
}

// ---------------------------------------------------------------------------
// lambda head:  lam[b] = sigmoid( relu_in[b,:256] . w + bias )
// 256 threads = 8 waves, one row per wave.
// ---------------------------------------------------------------------------
__global__ void lam2_kernel(const float* __restrict__ in, const float* __restrict__ w,
                            const float* __restrict__ b, float* __restrict__ lam)
{
    const int row  = blockIdx.x * 8 + (threadIdx.x >> 5);
    const int lane = threadIdx.x & 31;
    float s = 0.f;
    for (int k = lane; k < 256; k += 32) s += in[(size_t)row * 256 + k] * w[k];
    for (int off = 16; off > 0; off >>= 1) s += __shfl_down(s, off, 32);
    if (lane == 0) lam[row] = 1.f / (1.f + expf(-(s + b[0])));
}

// ---------------------------------------------------------------------------
// Differential attention over 2 tokens x 4 heads of 256 dims. One block / batch.
// Q/K/V buffers are [8192,1024] with row = b*2+t, col = h*256+d.
// ---------------------------------------------------------------------------
__global__ void diff_attn_kernel(const float* __restrict__ Q1, const float* __restrict__ K1,
                                 const float* __restrict__ Q2, const float* __restrict__ K2,
                                 const float* __restrict__ V,  const float* __restrict__ lam,
                                 float* __restrict__ att)
{
    __shared__ float sc[32];          // [h][which][i][j] logits
    __shared__ float df[4][2][2];     // relu(a1 - lam*a2)
    const int b = blockIdx.x, tid = threadIdx.x;
    {
        const int combo = tid >> 3, l8 = tid & 7;
        const int h = combo >> 3, rem = combo & 7;
        const int which = rem >> 2, i2 = (rem >> 1) & 1, j2 = rem & 1;
        const float* Q = which ? Q2 : Q1;
        const float* K = which ? K2 : K1;
        const float* qp = Q + (size_t)(b * 2 + i2) * 1024 + h * 256;
        const float* kp = K + (size_t)(b * 2 + j2) * 1024 + h * 256;
        float s = 0.f;
        for (int d = l8; d < 256; d += 8) s += qp[d] * kp[d];
        s += __shfl_down(s, 4, 8);
        s += __shfl_down(s, 2, 8);
        s += __shfl_down(s, 1, 8);
        if (l8 == 0) sc[combo] = s * 0.0625f;   // 256^-0.5
    }
    __syncthreads();
    if (tid < 8) {
        const int h = tid >> 1, i2 = tid & 1;
        const float lamv = lam[b];
        float x0 = sc[h * 8 + i2 * 2 + 0], x1 = sc[h * 8 + i2 * 2 + 1];
        float m1 = fmaxf(x0, x1);
        float e0 = expf(x0 - m1), e1 = expf(x1 - m1);
        float in1 = 1.f / (e0 + e1);
        float y0 = sc[h * 8 + 4 + i2 * 2 + 0], y1 = sc[h * 8 + 4 + i2 * 2 + 1];
        float m2 = fmaxf(y0, y1);
        float f0 = expf(y0 - m2), f1 = expf(y1 - m2);
        float in2 = 1.f / (f0 + f1);
        df[h][i2][0] = fmaxf(e0 * in1 - lamv * f0 * in2, 0.f);
        df[h][i2][1] = fmaxf(e1 * in1 - lamv * f1 * in2, 0.f);
    }
    __syncthreads();
    const float* v0 = V + (size_t)(b * 2) * 1024;
    const float* v1 = v0 + 1024;
    for (int idx = tid; idx < 2048; idx += 256) {
        const int t = idx >> 10, col = idx & 1023, h = col >> 8;
        att[(size_t)(b * 2 + t) * 1024 + col] = df[h][t][0] * v0[col] + df[h][t][1] * v1[col];
    }
}

// ---------------------------------------------------------------------------
// Small f32 linear for tiny N (dom l2: N=64, dom l3: N=7). One thread / output.
// ---------------------------------------------------------------------------
__global__ void small_linear_kernel(const float* __restrict__ A, int lda,
                                    const float* __restrict__ W, const float* __restrict__ bias,
                                    float* __restrict__ Y, int M, int N, int K, int relu)
{
    const long idx = (long)blockIdx.x * 256 + threadIdx.x;
    if (idx >= (long)M * N) return;
    const int m = (int)(idx / N), n = (int)(idx % N);
    float s = bias[n];
    for (int k = 0; k < K; k++) s += A[(size_t)m * lda + k] * W[(size_t)n * K + k];
    if (relu) s = fmaxf(s, 0.f);
    Y[idx] = s;
}

// ---------------------------------------------------------------------------
// |centroid_j|^2 precompute (7 blocks)
// ---------------------------------------------------------------------------
__global__ void centnorm_kernel(const float* __restrict__ cent, float* __restrict__ cn)
{
    __shared__ float red[256];
    const int j = blockIdx.x, tid = threadIdx.x;
    float a = 0.f;
    for (int d = tid; d < 1024; d += 256) { float v = cent[(size_t)j * 1024 + d]; a += v * v; }
    red[tid] = a; __syncthreads();
    for (int s = 128; s > 0; s >>= 1) { if (tid < s) red[tid] += red[tid + s]; __syncthreads(); }
    if (tid == 0) cn[j] = red[0];
}

// ---------------------------------------------------------------------------
// Final: centroid distances + argmax, mix, and concat into out [4096,4103].
// cs = [c | s] [4096,2048], hrec [4096,2048], dom [4096,7].
// ---------------------------------------------------------------------------
__global__ void finalize_kernel(const float* __restrict__ cs, const float* __restrict__ hrec,
                                const float* __restrict__ dom, const float* __restrict__ cent,
                                const float* __restrict__ counts, const float* __restrict__ cn,
                                float* __restrict__ out)
{
    __shared__ float red[256];
    __shared__ float vals[8];
    __shared__ int s_idx, s_mix;
    const int row = blockIdx.x, tid = threadIdx.x;
    const float* c = cs + (size_t)row * 2048;
    const float* s = c + 1024;

    float a[8] = {0.f,0.f,0.f,0.f,0.f,0.f,0.f,0.f};
    for (int d = tid; d < 1024; d += 256) {
        const float cv = c[d];
        a[0] += cv * cv;
        #pragma unroll
        for (int j = 0; j < 7; j++) a[1 + j] += cv * cent[(size_t)j * 1024 + d];
    }
    for (int q = 0; q < 8; q++) {
        red[tid] = a[q]; __syncthreads();
        for (int st = 128; st > 0; st >>= 1) { if (tid < st) red[tid] += red[tid + st]; __syncthreads(); }
        if (tid == 0) vals[q] = red[0];
        __syncthreads();
    }
    if (tid == 0) {
        const float c2 = vals[0];
        float best = -1.f; int bi = 0;
        for (int j = 0; j < 7; j++) {
            float d2 = c2 + cn[j] - 2.f * vals[1 + j];
            float dist = sqrtf(fmaxf(d2, 0.f));
            if (dist > best) { best = dist; bi = j; }
        }
        s_idx = bi;
        float csum = 0.f;
        for (int j = 0; j < 7; j++) csum += counts[j];
        s_mix = (csum > 0.f) ? 1 : 0;
    }
    __syncthreads();
    const int idx = s_idx, domix = s_mix;
    float* o = out + (size_t)row * 4103;
    const float* hc = cent + (size_t)idx * 1024;
    for (int d = tid; d < 1024; d += 256) {
        const float cv = c[d];
        o[d]        = domix ? (cv + 0.3f * (hc[d] - cv)) : cv;
        o[1024 + d] = s[d];
    }
    const float* hr = hrec + (size_t)row * 2048;
    for (int d = tid; d < 2048; d += 256) o[2048 + d] = hr[d];
    if (tid < 7) o[4096 + tid] = dom[(size_t)row * 7 + tid];
}

// ---------------------------------------------------------------------------
// Host orchestration
// ---------------------------------------------------------------------------
extern "C" void kernel_launch(void* const* d_in, const int* in_sizes, int n_in,
                              void* d_out, int out_size, void* d_ws, size_t ws_size,
                              hipStream_t stream)
{
    const int B = 4096, D = 1024;
    const int M2 = 8192;           // token rows for diff attention GEMMs

    // ---- unpack inputs in setup_inputs() dict order ----
    int ii = 0;
    const float* f_I = (const float*)d_in[ii++];
    const float* f_T = (const float*)d_in[ii++];
    // sa_I
    const float* saI_l1w = (const float*)d_in[ii++]; const float* saI_l1b = (const float*)d_in[ii++];
    const float* saI_lng = (const float*)d_in[ii++]; const float* saI_lnb = (const float*)d_in[ii++];
    const float* saI_l2w = (const float*)d_in[ii++]; const float* saI_l2b = (const float*)d_in[ii++];
    // sa_T
    const float* saT_l1w = (const float*)d_in[ii++]; const float* saT_l1b = (const float*)d_in[ii++];
    const float* saT_lng = (const float*)d_in[ii++]; const float* saT_lnb = (const float*)d_in[ii++];
    const float* saT_l2w = (const float*)d_in[ii++]; const float* saT_l2b = (const float*)d_in[ii++];
    // gca
    const float* vIw = (const float*)d_in[ii++]; const float* vIb = (const float*)d_in[ii++];
    const float* oIw = (const float*)d_in[ii++]; const float* oIb = (const float*)d_in[ii++];
    const float* vTw = (const float*)d_in[ii++]; const float* vTb = (const float*)d_in[ii++];
    const float* oTw = (const float*)d_in[ii++]; const float* oTb = (const float*)d_in[ii++];
    const float* lnIg = (const float*)d_in[ii++]; const float* lnIb = (const float*)d_in[ii++];
    const float* lnTg = (const float*)d_in[ii++]; const float* lnTb = (const float*)d_in[ii++];
    // diff
    const float* q1w = (const float*)d_in[ii++]; const float* q1b = (const float*)d_in[ii++];
    const float* k1w = (const float*)d_in[ii++]; const float* k1b = (const float*)d_in[ii++];
    const float* q2w = (const float*)d_in[ii++]; const float* q2b = (const float*)d_in[ii++];
    const float* k2w = (const float*)d_in[ii++]; const float* k2b = (const float*)d_in[ii++];
    const float* vw  = (const float*)d_in[ii++]; const float* vb  = (const float*)d_in[ii++];
    const float* ow  = (const float*)d_in[ii++]; const float* ob  = (const float*)d_in[ii++];
    const float* lam1w = (const float*)d_in[ii++]; const float* lam1b = (const float*)d_in[ii++];
    const float* lam2w = (const float*)d_in[ii++]; const float* lam2b = (const float*)d_in[ii++];
    // dis
    const float* c1w = (const float*)d_in[ii++]; const float* c1b = (const float*)d_in[ii++];
    const float* cln1g = (const float*)d_in[ii++]; const float* cln1b = (const float*)d_in[ii++];
    const float* c2w = (const float*)d_in[ii++]; const float* c2b = (const float*)d_in[ii++];
    const float* cln2g = (const float*)d_in[ii++]; const float* cln2b = (const float*)d_in[ii++];
    const float* s1w = (const float*)d_in[ii++]; const float* s1b = (const float*)d_in[ii++];
    const float* sln1g = (const float*)d_in[ii++]; const float* sln1b = (const float*)d_in[ii++];
    const float* s2w = (const float*)d_in[ii++]; const float* s2b = (const float*)d_in[ii++];
    const float* sln2g = (const float*)d_in[ii++]; const float* sln2b = (const float*)d_in[ii++];
    const float* d1w = (const float*)d_in[ii++]; const float* d1b = (const float*)d_in[ii++];
    const float* d2w = (const float*)d_in[ii++]; const float* d2b = (const float*)d_in[ii++];
    // dom
    const float* ml1w = (const float*)d_in[ii++]; const float* ml1b = (const float*)d_in[ii++];
    const float* ml2w = (const float*)d_in[ii++]; const float* ml2b = (const float*)d_in[ii++];
    const float* ml3w = (const float*)d_in[ii++]; const float* ml3b = (const float*)d_in[ii++];
    const float* centroids = (const float*)d_in[ii++];
    const float* counts    = (const float*)d_in[ii++];

    // ---- workspace bump allocator ----
    char* wsb = (char*)d_ws;
    size_t off = 0;
    auto alloc = [&](size_t bytes) -> void* {
        void* p = wsb + off;
        off = (off + bytes + 255) & ~(size_t)255;
        return p;
    };

    float* t1    = (float*)alloc((size_t)B * 256 * 4);
    float* t1g   = (float*)alloc((size_t)B * 256 * 4);
    float* fIr   = (float*)alloc((size_t)B * D * 4);
    float* fTr   = (float*)alloc((size_t)B * D * 4);
    float* tmpV  = (float*)alloc((size_t)B * D * 4);
    float* tmpO  = (float*)alloc((size_t)B * D * 4);
    float* fused = (float*)alloc((size_t)B * 2048 * 4);
    float* Q1    = (float*)alloc((size_t)M2 * D * 4);
    float* K1b_  = (float*)alloc((size_t)M2 * D * 4);
    float* Q2    = (float*)alloc((size_t)M2 * D * 4);
    float* K2b_  = (float*)alloc((size_t)M2 * D * 4);
    float* Vb_   = (float*)alloc((size_t)M2 * D * 4);
    float* lamr  = (float*)alloc((size_t)B * 256 * 4);
    float* lamv  = (float*)alloc((size_t)B * 4);
    float* att   = (float*)alloc((size_t)M2 * D * 4);
    float* x2    = (float*)alloc((size_t)B * 2048 * 4);
    float* big1  = (float*)alloc((size_t)B * 2048 * 4);
    float* cmid  = (float*)alloc((size_t)B * 1024 * 4);
    float* csb   = (float*)alloc((size_t)B * 2048 * 4);
    float* hrec  = (float*)alloc((size_t)B * 2048 * 4);
    float* dm1   = (float*)alloc((size_t)B * 128 * 4);
    float* dm2   = (float*)alloc((size_t)B * 64 * 4);
    float* dm3   = (float*)alloc((size_t)B * 7 * 4);
    float* cn    = (float*)alloc(7 * 4);

    // bf16 weight buffers
    auto balloc = [&](size_t n) -> __bf16* { return (__bf16*)alloc(n * 2); };
    __bf16* bsaI1 = balloc(256 * 1024);   __bf16* bsaI2 = balloc(1024 * 256);
    __bf16* bsaT1 = balloc(256 * 1024);   __bf16* bsaT2 = balloc(1024 * 256);
    __bf16* bvI = balloc(1024 * 1024);    __bf16* boI = balloc(1024 * 1024);
    __bf16* bvT = balloc(1024 * 1024);    __bf16* boT = balloc(1024 * 1024);
    __bf16* bq1 = balloc(1024 * 1024);    __bf16* bk1 = balloc(1024 * 1024);
    __bf16* bq2 = balloc(1024 * 1024);    __bf16* bk2 = balloc(1024 * 1024);
    __bf16* bv  = balloc(1024 * 1024);    __bf16* bo  = balloc(1024 * 1024);
    __bf16* blam1 = balloc(256 * 2048);
    __bf16* bc1 = balloc((size_t)2048 * 2048);  __bf16* bc2 = balloc((size_t)1024 * 2048);
    __bf16* bs1 = balloc((size_t)2048 * 2048);  __bf16* bs2 = balloc((size_t)1024 * 2048);
    __bf16* bd1 = balloc((size_t)2048 * 2048);  __bf16* bd2 = balloc((size_t)2048 * 2048);
    __bf16* bm1 = balloc(128 * 1024);

    auto cvt = [&](const float* src, __bf16* dst, long n) {
        cvt_bf16_kernel<<<dim3((unsigned)((n + 255) / 256)), dim3(256), 0, stream>>>(src, dst, n);
    };
    cvt(saI_l1w, bsaI1, 256 * 1024);  cvt(saI_l2w, bsaI2, 1024 * 256);
    cvt(saT_l1w, bsaT1, 256 * 1024);  cvt(saT_l2w, bsaT2, 1024 * 256);
    cvt(vIw, bvI, 1024 * 1024);  cvt(oIw, boI, 1024 * 1024);
    cvt(vTw, bvT, 1024 * 1024);  cvt(oTw, boT, 1024 * 1024);
    cvt(q1w, bq1, 1024 * 1024);  cvt(k1w, bk1, 1024 * 1024);
    cvt(q2w, bq2, 1024 * 1024);  cvt(k2w, bk2, 1024 * 1024);
    cvt(vw,  bv,  1024 * 1024);  cvt(ow,  bo,  1024 * 1024);
    cvt(lam1w, blam1, 256 * 2048);
    cvt(c1w, bc1, (long)2048 * 2048);  cvt(c2w, bc2, (long)1024 * 2048);
    cvt(s1w, bs1, (long)2048 * 2048);  cvt(s2w, bs2, (long)1024 * 2048);
    cvt(d1w, bd1, (long)2048 * 2048);  cvt(d2w, bd2, (long)2048 * 2048);
    cvt(ml1w, bm1, 128 * 1024);

    // dispatch: N%256==0 -> 128x256 tiles; N==128 (dom.l1) -> 256x128 tiles
    auto gemm = [&](const float* A, long lda, const __bf16* Wb, const float* bias,
                    const float* res, float* Yp, int M, int N, int K, int act) {
        if (N % 256 == 0) {
            dim3 g((unsigned)(N / 256), (unsigned)(M / 128));
            if (res)
                gemm_bf16_kernel<128,256,0,true ><<<g, 256, 0, stream>>>(A, lda, Wb, bias, res, Yp, M, N, K);
            else if (act == 1)
                gemm_bf16_kernel<128,256,1,false><<<g, 256, 0, stream>>>(A, lda, Wb, bias, nullptr, Yp, M, N, K);
            else if (act == 2)
                gemm_bf16_kernel<128,256,2,false><<<g, 256, 0, stream>>>(A, lda, Wb, bias, nullptr, Yp, M, N, K);
            else
                gemm_bf16_kernel<128,256,0,false><<<g, 256, 0, stream>>>(A, lda, Wb, bias, nullptr, Yp, M, N, K);
        } else {
            dim3 g((unsigned)(N / 128), (unsigned)(M / 256));
            gemm_bf16_kernel<256,128,1,false><<<g, 256, 0, stream>>>(A, lda, Wb, bias, nullptr, Yp, M, N, K);
        }
    };
    auto ln = [&](const float* in, long ip, const float* g, const float* b,
                  float* out, long op, int d, int gelu_flag) {
        size_t sm = (size_t)(d + 256) * sizeof(float);
        layernorm_kernel<<<dim3(B), dim3(256), sm, stream>>>(in, ip, g, b, out, op, d, gelu_flag);
    };

    // ---- self-refine I ----
    gemm(f_I, 1024, bsaI1, saI_l1b, nullptr, t1, B, 256, 1024, 0);
    ln(t1, 256, saI_lng, saI_lnb, t1g, 256, 256, 1);
    gemm(t1g, 256, bsaI2, saI_l2b, f_I, fIr, B, 1024, 256, 0);
    // ---- self-refine T ----
    gemm(f_T, 1024, bsaT1, saT_l1b, nullptr, t1, B, 256, 1024, 0);
    ln(t1, 256, saT_lng, saT_lnb, t1g, 256, 256, 1);
    gemm(t1g, 256, bsaT2, saT_l2b, f_T, fTr, B, 1024, 256, 0);

    // ---- gated cross attention ----
    gemm(fTr, 1024, bvI, vIb, nullptr, tmpV, B, 1024, 1024, 0);
    gemm(tmpV, 1024, boI, oIb, fIr, tmpO, B, 1024, 1024, 0);
    ln(tmpO, 1024, lnIg, lnIb, fused, 2048, 1024, 0);             // -> fused[:, 0:1024]
    gemm(fIr, 1024, bvT, vTb, nullptr, tmpV, B, 1024, 1024, 0);
    gemm(tmpV, 1024, boT, oTb, fTr, tmpO, B, 1024, 1024, 0);
    ln(tmpO, 1024, lnTg, lnTb, fused + 1024, 2048, 1024, 0);      // -> fused[:, 1024:2048]

    // ---- differential attention ----
    // tokens view: fused [4096,2048] == [8192,1024]
    gemm(fused, 1024, bq1, q1b, nullptr, Q1,   M2, 1024, 1024, 0);
    gemm(fused, 1024, bk1, k1b, nullptr, K1b_, M2, 1024, 1024, 0);
    gemm(fused, 1024, bq2, q2b, nullptr, Q2,   M2, 1024, 1024, 0);
    gemm(fused, 1024, bk2, k2b, nullptr, K2b_, M2, 1024, 1024, 0);
    gemm(fused, 1024, bv,  vb,  nullptr, Vb_,  M2, 1024, 1024, 0);
    gemm(fused, 2048, blam1, lam1b, nullptr, lamr, B, 256, 2048, 1);      // relu fused
    lam2_kernel<<<dim3(B / 8), dim3(256), 0, stream>>>(lamr, lam2w, lam2b, lamv);
    diff_attn_kernel<<<dim3(B), dim3(256), 0, stream>>>(Q1, K1b_, Q2, K2b_, Vb_, lamv, att);
    gemm(att, 1024, bo, ob, fused, x2, M2, 1024, 1024, 0);                // + tokens residual

    // ---- disentangle: content branch ----
    gemm(x2, 2048, bc1, c1b, nullptr, big1, B, 2048, 2048, 0);
    ln(big1, 2048, cln1g, cln1b, big1, 2048, 2048, 1);                    // gelu(LN(.))
    gemm(big1, 2048, bc2, c2b, nullptr, cmid, B, 1024, 2048, 0);
    ln(cmid, 1024, cln2g, cln2b, csb, 2048, 1024, 0);                     // c -> cs[:,0:1024]
    // ---- style branch ----
    gemm(x2, 2048, bs1, s1b, nullptr, big1, B, 2048, 2048, 0);
    ln(big1, 2048, sln1g, sln1b, big1, 2048, 2048, 1);
    gemm(big1, 2048, bs2, s2b, nullptr, cmid, B, 1024, 2048, 0);
    ln(cmid, 1024, sln2g, sln2b, csb + 1024, 2048, 1024, 0);              // s -> cs[:,1024:2048]
    // ---- reconstruction ----
    gemm(csb, 2048, bd1, d1b, nullptr, big1, B, 2048, 2048, 2);           // gelu fused
    gemm(big1, 2048, bd2, d2b, nullptr, hrec, B, 2048, 2048, 0);

    // ---- domain head (from s) ----
    gemm(csb + 1024, 2048, bm1, ml1b, nullptr, dm1, B, 128, 1024, 1);     // relu fused (256x128 path)
    small_linear_kernel<<<dim3((B * 64 + 255) / 256), dim3(256), 0, stream>>>(
        dm1, 128, ml2w, ml2b, dm2, B, 64, 128, 1);
    small_linear_kernel<<<dim3((B * 7 + 255) / 256), dim3(256), 0, stream>>>(
        dm2, 64, ml3w, ml3b, dm3, B, 7, 64, 0);

    // ---- centroid distances + final concat ----
    centnorm_kernel<<<dim3(7), dim3(256), 0, stream>>>(centroids, cn);
    finalize_kernel<<<dim3(B), dim3(256), 0, stream>>>(
        csb, hrec, dm3, centroids, counts, cn, (float*)d_out);
}
